// FFTMultiHeadSelfAttention_31086973289253
// MI455X (gfx1250) — matrix operand order repaired and checked by reading, not compiled, not run on hardware
//
#include <hip/hip_runtime.h>

#define D_MODEL 1024
#define NH 16
#define DH 64
#define SEQ 2048
#define BATCHN 4
#define MTOT (BATCHN * SEQ)
#define K_STEP 32
#define LDPAD 40     // halves per LDS row for 32-wide K tiles (80B, 16B aligned)
#define LDK 72       // halves per LDS row for 64-wide tiles (144B, 16B aligned)

typedef __attribute__((ext_vector_type(16))) __bf16 v16bf;
typedef __attribute__((ext_vector_type(8)))  float  v8f;
typedef unsigned int u32x4 __attribute__((ext_vector_type(4)));
typedef int          i32x4 __attribute__((ext_vector_type(4)));
typedef int          i32x8 __attribute__((ext_vector_type(8)));

struct __align__(16) U128 { unsigned int x0, x1, x2, x3; };
struct __align__(16) F128 { float x0, x1, x2, x3; };

union BF16x16 {
    v16bf v;
    unsigned int u[8];
    unsigned short h[16];
};

__device__ inline unsigned short f32_to_bf16_rn(float f) {
    unsigned int u = __float_as_uint(f);
    u += 0x7FFFu + ((u >> 16) & 1u);
    return (unsigned short)(u >> 16);
}

__device__ inline unsigned int pack_bf16x2(float a, float b) {
#if __has_builtin(__builtin_amdgcn_cvt_pk_bf16_f32)
    auto r = __builtin_amdgcn_cvt_pk_bf16_f32(a, b);
    unsigned int u = 0;
    __builtin_memcpy(&u, &r, sizeof(r) < sizeof(u) ? sizeof(r) : sizeof(u));
    return u;
#else
    return (unsigned int)f32_to_bf16_rn(a) | ((unsigned int)f32_to_bf16_rn(b) << 16);
#endif
}

__device__ inline v8f zero8() {
    v8f z;
    #pragma unroll
    for (int i = 0; i < 8; i++) z[i] = 0.0f;
    return z;
}

__device__ inline void frag_set(BF16x16& f, U128 a, U128 b) {
    f.u[0] = a.x0; f.u[1] = a.x1; f.u[2] = a.x2; f.u[3] = a.x3;
    f.u[4] = b.x0; f.u[5] = b.x1; f.u[6] = b.x2; f.u[7] = b.x3;
}

__device__ inline v8f wmma_bf16(const BF16x16& a, const BF16x16& b, v8f c) {
    return __builtin_amdgcn_wmma_f32_16x16x32_bf16(
        /*neg_a=*/false, a.v, /*neg_b=*/false, b.v,
        /*c_mod=*/(short)0, c, /*reuse_a=*/false, /*reuse_b=*/false);
}

// -------------------------------------------------------------------------
// TDM: load a 2D tile (tile_d1 rows x tile_d0 elements, 2-byte elements)
// from a row-major tensor (row length tensor_d0, row stride stride0, tensor_d1
// rows) into LDS at lds_addr.  LDS padding: +4 dwords after every 32 dwords
// (=> 64-half rows land on a 72-half LDS pitch, matching LDK).
// D# layout per CDNA5 ISA.
// -------------------------------------------------------------------------
__device__ inline void tdm_load_2d_bf16(unsigned int lds_addr, const void* gaddr,
                                        unsigned int tensor_d0, unsigned int tensor_d1,
                                        unsigned int stride0,
                                        unsigned int tile_d0, unsigned int tile_d1)
{
    unsigned long long ga = (unsigned long long)gaddr;
    u32x4 g0;
    g0[0] = 1u;                                   // count=1 (valid user descriptor)
    g0[1] = lds_addr;                             // lds_addr @ bits 63:32
    g0[2] = (unsigned int)ga;                     // global_addr[31:0] @ bits 95:64
    g0[3] = (unsigned int)((ga >> 32) & 0x01FFFFFFull) | (2u << 30); // addr[56:32], type=2
    i32x8 g1;
    // data_size=1 (2B) @16, pad_enable @20, pad_interval=4 (32 dwords) @22,
    // pad_amount=3 (4 dwords) @25
    g1[0] = (int)((1u << 16) | (1u << 20) | (4u << 22) | (3u << 25));
    g1[1] = (int)(tensor_d0 << 16);                          // tensor_dim0[15:0]
    g1[2] = (int)((tensor_d0 >> 16) | (tensor_d1 << 16));    // dim0 hi | dim1 lo
    g1[3] = (int)((tensor_d1 >> 16) | (tile_d0 << 16));      // dim1 hi | tile_dim0
    g1[4] = (int)(tile_d1 & 0xFFFFu);                        // tile_dim1, tile_dim2=0
    g1[5] = (int)stride0;                                    // tensor_dim0_stride[31:0]
    g1[6] = 0;                                               // stride0 hi, dim1_stride=0
    g1[7] = 0;
    i32x4 z4 = (i32x4)0;
#if defined(__clang_major__) && (__clang_major__ >= 23)
    i32x8 z8 = (i32x8)0;
    __builtin_amdgcn_tensor_load_to_lds(g0, g1, z4, z4, z8, 0);
#else
    __builtin_amdgcn_tensor_load_to_lds(g0, g1, z4, z4, 0);
#endif
}

// -------------------------------------------------------------------------
// Kernel 1: QKV projection.  C[m,n] = sum_k X[m,k]*W[n,k] + bias[n]
// Workgroup tile: M=64, N=128, 8 waves (4 along M x 2 along N), K step 32.
// Q,K stored bf16 [B,H,S,Dh] (Q pre-scaled by 1/sqrt(DH)); V stored bf16
// TRANSPOSED [B,H,Dh,S] so the attention kernel can TDM-load it directly.
// -------------------------------------------------------------------------
__global__ __launch_bounds__(256)
void qkv_proj_kernel(const float* __restrict__ X,
                     const float* __restrict__ Wq, const float* __restrict__ bq,
                     const float* __restrict__ Wk, const float* __restrict__ bk,
                     const float* __restrict__ Wv, const float* __restrict__ bv,
                     unsigned short* __restrict__ q_ws,
                     unsigned short* __restrict__ k_ws,
                     unsigned short* __restrict__ v_ws)
{
    __shared__ unsigned short lA[64 * LDPAD];
    __shared__ unsigned short lB[128 * LDPAD];

    const int tid  = threadIdx.x;
    const int wave = tid >> 5;
    const int lane = tid & 31;
    const int g    = lane >> 4;
    const int ln   = lane & 15;
    const int wm   = wave & 3;
    const int wn   = wave >> 2;

    const int m0 = blockIdx.x * 64;
    const int n0 = blockIdx.y * 128;
    const int which = blockIdx.z;

    const float* W    = (which == 0) ? Wq : (which == 1) ? Wk : Wv;
    const float* bias = (which == 0) ? bq : (which == 1) ? bk : bv;
    unsigned short* out = (which == 0) ? q_ws : (which == 1) ? k_ws : v_ws;
    const float oscale = (which == 0) ? 0.125f : 1.0f;   // 1/sqrt(64) folded into Q
    const bool vtrans = (which == 2);

    v8f acc[4];
    #pragma unroll
    for (int j = 0; j < 4; j++) acc[j] = zero8();

    const int arow = tid >> 2;           // A stage: 0..63
    const int akk  = (tid & 3) * 8;
    const int brow = tid >> 1;           // B stage: 0..127
    const int bkk  = (tid & 1) * 16;

    for (int k0 = 0; k0 < D_MODEL; k0 += K_STEP) {
        // ---- stage A: 64x32 f32 -> bf16 ----
        {
            const float* src = X + (size_t)(m0 + arow) * D_MODEL + k0 + akk;
            __builtin_prefetch(src + K_STEP, 0, 0);
            F128 f0 = *(const F128*)src;
            F128 f1 = *(const F128*)(src + 4);
            U128 p;
            p.x0 = pack_bf16x2(f0.x0, f0.x1); p.x1 = pack_bf16x2(f0.x2, f0.x3);
            p.x2 = pack_bf16x2(f1.x0, f1.x1); p.x3 = pack_bf16x2(f1.x2, f1.x3);
            *(U128*)&lA[arow * LDPAD + akk] = p;
        }
        // ---- stage B: 128x32 f32 -> bf16 ----
        {
            const float* src = W + (size_t)(n0 + brow) * D_MODEL + k0 + bkk;
            __builtin_prefetch(src + K_STEP, 0, 0);
            F128 f0 = *(const F128*)src;
            F128 f1 = *(const F128*)(src + 4);
            F128 f2 = *(const F128*)(src + 8);
            F128 f3 = *(const F128*)(src + 12);
            U128 p0, p1;
            p0.x0 = pack_bf16x2(f0.x0, f0.x1); p0.x1 = pack_bf16x2(f0.x2, f0.x3);
            p0.x2 = pack_bf16x2(f1.x0, f1.x1); p0.x3 = pack_bf16x2(f1.x2, f1.x3);
            p1.x0 = pack_bf16x2(f2.x0, f2.x1); p1.x1 = pack_bf16x2(f2.x2, f2.x3);
            p1.x2 = pack_bf16x2(f3.x0, f3.x1); p1.x3 = pack_bf16x2(f3.x2, f3.x3);
            *(U128*)&lB[brow * LDPAD + bkk]     = p0;
            *(U128*)&lB[brow * LDPAD + bkk + 8] = p1;
        }
        __syncthreads();

        // ---- compute ----
        BF16x16 a;
        {
            const unsigned short* pa = &lA[(wm * 16 + ln) * LDPAD];
            U128 t0 = *(const U128*)(pa + g * 8);
            U128 t1 = *(const U128*)(pa + 16 + g * 8);
            frag_set(a, t0, t1);
        }
        #pragma unroll
        for (int j = 0; j < 4; j++) {
            BF16x16 b;
            const unsigned short* pb = &lB[(wn * 64 + j * 16 + ln) * LDPAD];
            U128 t0 = *(const U128*)(pb + g * 8);
            U128 t1 = *(const U128*)(pb + 16 + g * 8);
            frag_set(b, t0, t1);
            acc[j] = wmma_bf16(a, b, acc[j]);
        }
        __syncthreads();
    }

    // ---- epilogue: bias, scale, scatter ----
    #pragma unroll
    for (int j = 0; j < 4; j++) {
        const int n_g = n0 + wn * 64 + j * 16 + ln;
        const float bv_ = bias[n_g];
        const int h = n_g >> 6, d = n_g & 63;
        #pragma unroll
        for (int v = 0; v < 8; v++) {
            const int m_g = m0 + wm * 16 + g * 8 + v;
            const int bidx = m_g >> 11;          // / SEQ
            const int s    = m_g & (SEQ - 1);
            const float val = (acc[j][v] + bv_) * oscale;
            const size_t oidx = vtrans
                ? (((size_t)(bidx * NH + h) * DH + d) * SEQ + s)   // V: [B,H,Dh,S]
                : (((size_t)(bidx * NH + h) * SEQ + s) * DH + d);  // Q,K: [B,H,S,Dh]
            out[oidx] = f32_to_bf16_rn(val);
        }
    }
}

// -------------------------------------------------------------------------
// Kernel 2: flash attention per (b,h).  4 waves x 16 q rows, 64-key tiles.
// K and (pre-transposed) V tiles are staged LDS via the Tensor Data Mover,
// issued by wave 0, padded by TDM to the 72-half LDS pitch.
// -------------------------------------------------------------------------
__global__ __launch_bounds__(128)
void attn_kernel(const unsigned short* __restrict__ q_ws,
                 const unsigned short* __restrict__ k_ws,
                 const unsigned short* __restrict__ v_ws,   // [B,H,Dh,S]
                 unsigned short* __restrict__ attn_ws)
{
    __shared__ unsigned short lK[64 * LDK];        // [key][dh]
    __shared__ unsigned short lV[64 * LDK];        // [dh][key]
    __shared__ unsigned short lP[4][16 * LDK];     // per-wave P tile [m][key]

    const int tid  = threadIdx.x;
    const int wave = tid >> 5;
    const int lane = tid & 31;
    const int g    = lane >> 4;
    const int ln   = lane & 15;

    const int qt = blockIdx.x;
    const int bh = blockIdx.y;
    const size_t basebh = (size_t)bh * SEQ * DH;
    const int q0 = qt * 64 + wave * 16;

    const unsigned int lK_lds = (unsigned int)(size_t)&lK[0];
    const unsigned int lV_lds = (unsigned int)(size_t)&lV[0];

    // preload Q A-fragments (dh split into two K=32 chunks); Q already scaled
    BF16x16 qa[2];
    #pragma unroll
    for (int c = 0; c < 2; c++) {
        const unsigned short* src = q_ws + basebh + (size_t)(q0 + ln) * DH + c * 32 + g * 8;
        U128 t0 = *(const U128*)src;
        U128 t1 = *(const U128*)(src + 16);
        frag_set(qa[c], t0, t1);
    }

    float m_run[8], l_run[8];
    #pragma unroll
    for (int v = 0; v < 8; v++) { m_run[v] = -1e30f; l_run[v] = 0.0f; }
    v8f o[4];
    #pragma unroll
    for (int j = 0; j < 4; j++) o[j] = zero8();

    for (int kt = 0; kt < SEQ / 64; kt++) {
        // ---- TDM stage: K tile [64 keys][64 dh], V tile [64 dh][64 keys] ----
        if (wave == 0) {
            // K: tensor row len = DH, SEQ rows; tile 64x64 starting at key kt*64
            tdm_load_2d_bf16(lK_lds, k_ws + basebh + (size_t)(kt * 64) * DH,
                             DH, SEQ, DH, /*tile_d0=*/DH, /*tile_d1=*/64);
            // V^T: tensor row len = SEQ, DH rows; tile 64x64 starting at col kt*64
            tdm_load_2d_bf16(lV_lds, v_ws + basebh + (size_t)(kt * 64),
                             SEQ, DH, SEQ, /*tile_d0=*/64, /*tile_d1=*/DH);
            __builtin_amdgcn_s_wait_tensorcnt(0);
        }
        __syncthreads();

        // ---- scores S = Q K^T  (8 WMMA) ----
        v8f s[4];
        #pragma unroll
        for (int j = 0; j < 4; j++) {
            v8f sc = zero8();
            #pragma unroll
            for (int c = 0; c < 2; c++) {
                BF16x16 kb;
                const unsigned short* pb = &lK[(j * 16 + ln) * LDK + c * 32];
                U128 t0 = *(const U128*)(pb + g * 8);
                U128 t1 = *(const U128*)(pb + 16 + g * 8);
                frag_set(kb, t0, t1);
                sc = wmma_bf16(qa[c], kb, sc);
            }
            s[j] = sc;
        }

        // ---- online softmax over the 64-key tile ----
        #pragma unroll
        for (int v = 0; v < 8; v++) {
            float rmax = fmaxf(fmaxf(s[0][v], s[1][v]), fmaxf(s[2][v], s[3][v]));
            #pragma unroll
            for (int off = 1; off < 16; off <<= 1)
                rmax = fmaxf(rmax, __shfl_xor(rmax, off, 32));
            const float mnew  = fmaxf(m_run[v], rmax);
            const float alpha = __expf(m_run[v] - mnew);
            float psum = 0.0f;
            #pragma unroll
            for (int j = 0; j < 4; j++) {
                float p = __expf(s[j][v] - mnew);
                s[j][v] = p;
                psum += p;
            }
            #pragma unroll
            for (int off = 1; off < 16; off <<= 1)
                psum += __shfl_xor(psum, off, 32);
            l_run[v] = l_run[v] * alpha + psum;
            m_run[v] = mnew;
            #pragma unroll
            for (int j = 0; j < 4; j++) o[j][v] *= alpha;
        }

        // ---- P -> per-wave LDS (D-layout -> A-layout reshape) ----
        unsigned short* myP = &lP[wave][0];
        #pragma unroll
        for (int j = 0; j < 4; j++)
            #pragma unroll
            for (int v = 0; v < 8; v++)
                myP[(g * 8 + v) * LDK + j * 16 + ln] = f32_to_bf16_rn(s[j][v]);

        asm volatile("s_wait_dscnt 0" ::: "memory");  // same-wave LDS RAW

        // ---- O += P * V  (8 WMMA) ----
        #pragma unroll
        for (int j = 0; j < 4; j++) {
            #pragma unroll
            for (int c = 0; c < 2; c++) {
                BF16x16 pa, vb;
                const unsigned short* ppa = myP + ln * LDK + c * 32 + g * 8;
                U128 a0 = *(const U128*)ppa;
                U128 a1 = *(const U128*)(ppa + 16);
                frag_set(pa, a0, a1);
                const unsigned short* pvb = &lV[(j * 16 + ln) * LDK + c * 32];
                U128 b0 = *(const U128*)(pvb + g * 8);
                U128 b1 = *(const U128*)(pvb + 16 + g * 8);
                frag_set(vb, b0, b1);
                o[j] = wmma_bf16(pa, vb, o[j]);
            }
        }
        __syncthreads();   // all reads of lK/lV done before next TDM overwrite
    }

    // ---- epilogue: normalize, write attn [M, D] bf16 ----
    const int b = bh >> 4, h = bh & 15;
    #pragma unroll
    for (int j = 0; j < 4; j++) {
        #pragma unroll
        for (int v = 0; v < 8; v++) {
            const int row = q0 + g * 8 + v;
            const float val = o[j][v] / l_run[v];
            attn_ws[(size_t)(b * SEQ + row) * D_MODEL + h * 64 + j * 16 + ln] =
                f32_to_bf16_rn(val);
        }
    }
}

// -------------------------------------------------------------------------
// Kernel 3: output projection + bias + residual  -> f32 workspace
// -------------------------------------------------------------------------
__global__ __launch_bounds__(256)
void out_proj_kernel(const unsigned short* __restrict__ A,   // attn bf16 [M,D]
                     const float* __restrict__ W,            // wo [N,K]
                     const float* __restrict__ bias,
                     const float* __restrict__ X,            // residual f32
                     float* __restrict__ Y)
{
    __shared__ unsigned short lA[64 * LDPAD];
    __shared__ unsigned short lB[128 * LDPAD];

    const int tid  = threadIdx.x;
    const int wave = tid >> 5;
    const int lane = tid & 31;
    const int g    = lane >> 4;
    const int ln   = lane & 15;
    const int wm   = wave & 3;
    const int wn   = wave >> 2;

    const int m0 = blockIdx.x * 64;
    const int n0 = blockIdx.y * 128;

    v8f acc[4];
    #pragma unroll
    for (int j = 0; j < 4; j++) acc[j] = zero8();

    const int arow = tid >> 2;
    const int akk  = (tid & 3) * 8;
    const int brow = tid >> 1;
    const int bkk  = (tid & 1) * 16;

    for (int k0 = 0; k0 < D_MODEL; k0 += K_STEP) {
        {   // A already bf16 -> straight copy
            const unsigned short* src = A + (size_t)(m0 + arow) * D_MODEL + k0 + akk;
            __builtin_prefetch(src + K_STEP, 0, 0);
            *(U128*)&lA[arow * LDPAD + akk] = *(const U128*)src;
        }
        {
            const float* src = W + (size_t)(n0 + brow) * D_MODEL + k0 + bkk;
            __builtin_prefetch(src + K_STEP, 0, 0);
            F128 f0 = *(const F128*)src;
            F128 f1 = *(const F128*)(src + 4);
            F128 f2 = *(const F128*)(src + 8);
            F128 f3 = *(const F128*)(src + 12);
            U128 p0, p1;
            p0.x0 = pack_bf16x2(f0.x0, f0.x1); p0.x1 = pack_bf16x2(f0.x2, f0.x3);
            p0.x2 = pack_bf16x2(f1.x0, f1.x1); p0.x3 = pack_bf16x2(f1.x2, f1.x3);
            p1.x0 = pack_bf16x2(f2.x0, f2.x1); p1.x1 = pack_bf16x2(f2.x2, f2.x3);
            p1.x2 = pack_bf16x2(f3.x0, f3.x1); p1.x3 = pack_bf16x2(f3.x2, f3.x3);
            *(U128*)&lB[brow * LDPAD + bkk]     = p0;
            *(U128*)&lB[brow * LDPAD + bkk + 8] = p1;
        }
        __syncthreads();

        BF16x16 a;
        {
            const unsigned short* pa = &lA[(wm * 16 + ln) * LDPAD];
            U128 t0 = *(const U128*)(pa + g * 8);
            U128 t1 = *(const U128*)(pa + 16 + g * 8);
            frag_set(a, t0, t1);
        }
        #pragma unroll
        for (int j = 0; j < 4; j++) {
            BF16x16 b;
            const unsigned short* pb = &lB[(wn * 64 + j * 16 + ln) * LDPAD];
            U128 t0 = *(const U128*)(pb + g * 8);
            U128 t1 = *(const U128*)(pb + 16 + g * 8);
            frag_set(b, t0, t1);
            acc[j] = wmma_bf16(a, b, acc[j]);
        }
        __syncthreads();
    }

    #pragma unroll
    for (int j = 0; j < 4; j++) {
        const int n_g = n0 + wn * 64 + j * 16 + ln;
        const float bo_ = bias[n_g];
        #pragma unroll
        for (int v = 0; v < 8; v++) {
            const int m_g = m0 + wm * 16 + g * 8 + v;
            const size_t idx = (size_t)m_g * D_MODEL + n_g;
            Y[idx] = acc[j][v] + bo_ + X[idx];
        }
    }
}

// -------------------------------------------------------------------------
// Kernel 4: rowwise LayerNorm -> d_out (f32)
// -------------------------------------------------------------------------
__global__ __launch_bounds__(256)
void ln_kernel(const float* __restrict__ Y,
               const float* __restrict__ gamma,
               const float* __restrict__ beta,
               float* __restrict__ out)
{
    __shared__ float stage[8];
    __shared__ float sh_mu, sh_rstd;

    const int row = blockIdx.x;
    const int tid = threadIdx.x;

    F128 x = *(const F128*)(Y + (size_t)row * D_MODEL + tid * 4);

    float s = x.x0 + x.x1 + x.x2 + x.x3;
    #pragma unroll
    for (int off = 1; off < 32; off <<= 1) s += __shfl_xor(s, off, 32);
    if ((tid & 31) == 0) stage[tid >> 5] = s;
    __syncthreads();
    if (tid == 0) {
        float t = 0.0f;
        for (int i = 0; i < 8; i++) t += stage[i];
        sh_mu = t * (1.0f / D_MODEL);
    }
    __syncthreads();
    const float mu = sh_mu;

    const float d0 = x.x0 - mu, d1 = x.x1 - mu, d2 = x.x2 - mu, d3 = x.x3 - mu;
    float ss = d0 * d0 + d1 * d1 + d2 * d2 + d3 * d3;
    #pragma unroll
    for (int off = 1; off < 32; off <<= 1) ss += __shfl_xor(ss, off, 32);
    if ((tid & 31) == 0) stage[tid >> 5] = ss;
    __syncthreads();
    if (tid == 0) {
        float t = 0.0f;
        for (int i = 0; i < 8; i++) t += stage[i];
        sh_rstd = rsqrtf(t * (1.0f / D_MODEL) + 1e-5f);
    }
    __syncthreads();
    const float r = sh_rstd;

    const int c = tid * 4;
    F128 gg = *(const F128*)(gamma + c);
    F128 bb = *(const F128*)(beta + c);
    F128 o;
    o.x0 = d0 * r * gg.x0 + bb.x0;
    o.x1 = d1 * r * gg.x1 + bb.x1;
    o.x2 = d2 * r * gg.x2 + bb.x2;
    o.x3 = d3 * r * gg.x3 + bb.x3;
    *(F128*)(out + (size_t)row * D_MODEL + c) = o;
}

// -------------------------------------------------------------------------
extern "C" void kernel_launch(void* const* d_in, const int* in_sizes, int n_in,
                              void* d_out, int out_size, void* d_ws, size_t ws_size,
                              hipStream_t stream) {
    const float* batch = (const float*)d_in[0];
    const float* wq = (const float*)d_in[1];
    const float* bq = (const float*)d_in[2];
    const float* wk = (const float*)d_in[3];
    const float* bk = (const float*)d_in[4];
    const float* wv = (const float*)d_in[5];
    const float* bv = (const float*)d_in[6];
    const float* wo = (const float*)d_in[7];
    const float* bo = (const float*)d_in[8];
    const float* ln_g = (const float*)d_in[9];
    const float* ln_b = (const float*)d_in[10];

    const size_t mat = (size_t)MTOT * D_MODEL;
    unsigned short* q_ws    = (unsigned short*)d_ws;
    unsigned short* k_ws    = q_ws + mat;
    unsigned short* v_ws    = k_ws + mat;
    unsigned short* attn_ws = v_ws + mat;
    float*          y_ws    = (float*)(attn_ws + mat);

    qkv_proj_kernel<<<dim3(MTOT / 64, D_MODEL / 128, 3), 256, 0, stream>>>(
        batch, wq, bq, wk, bk, wv, bv, q_ws, k_ws, v_ws);

    attn_kernel<<<dim3(SEQ / 64, BATCHN * NH), 128, 0, stream>>>(
        q_ws, k_ws, v_ws, attn_ws);

    out_proj_kernel<<<dim3(MTOT / 64, D_MODEL / 128), 256, 0, stream>>>(
        attn_ws, wo, bo, batch, y_ws);

    ln_kernel<<<dim3(MTOT), 256, 0, stream>>>(y_ws, ln_g, ln_b, (float*)d_out);
}
